// RNNCellModel_54443005444541
// MI455X (gfx1250) — matrix operand order repaired and checked by reading, not compiled, not run on hardware
//
#include <hip/hip_runtime.h>
#include <stdint.h>

// ---------------------------------------------------------------------------
// Cortical 4-population tanh RNN, T=50 sequential steps, B=16.
// Memory-bound on weights (1.36GB fp32 per step). Strategy:
//   * one-time repack of all weights to bf16 in WMMA-B-fragment tile layout
//   * per step: gather concat-activations (bf16) + WMMA bf16 GEMM + tanh
//   * v_wmma_f32_16x16x32_bf16, M=16 == batch, fp32 accumulation
// ---------------------------------------------------------------------------

#define BB 16
#define TT 50

typedef __bf16 bf16_t;
typedef bf16_t   v16bf __attribute__((ext_vector_type(16)));
typedef float    v8f   __attribute__((ext_vector_type(8)));
typedef uint32_t u32x4 __attribute__((ext_vector_type(4)));

union Frag16 { v16bf v; u32x4 q[2]; };

__device__ __forceinline__ uint16_t f32_to_bf16_rne(float f) {
    uint32_t u = __float_as_uint(f);
    uint32_t r = u + 0x7FFFu + ((u >> 16) & 1u);   // round-to-nearest-even
    return (uint16_t)(r >> 16);
}

// ---------------------------------------------------------------------------
// Pack [wih | whh] (fp32, row-major over N rows) into bf16 WMMA-B tiles:
//   dst[nt][kc][lane][e]  with  n = nt*16 + (lane&15)
//                               k = kc*32 + ((lane>>4)<<3) + (e&7) + ((e>>3)<<4)
// (lanes 0-15 hold K={0..7,16..23} of the 32-chunk, lanes 16-31 hold
//  K={8..15,24..31}, matching the A fragment). K zero-padded to Kpad.
// ---------------------------------------------------------------------------
__global__ void pack_weights_bf16(const float* __restrict__ wih, int Kin,
                                  const float* __restrict__ whh, int Nhh,
                                  int N, int Kpad, uint16_t* __restrict__ dst) {
    long long idx   = (long long)blockIdx.x * blockDim.x + threadIdx.x;
    long long total = (long long)N * Kpad;
    if (idx >= total) return;
    int e    = (int)(idx & 15);
    int lane = (int)((idx >> 4) & 31);
    long long grp = idx >> 9;
    int Kc = Kpad >> 5;
    int kc = (int)(grp % Kc);
    int nt = (int)(grp / Kc);
    int n  = nt * 16 + (lane & 15);
    int k  = kc * 32 + ((lane >> 4) << 3) + (e & 7) + ((e >> 3) << 4);
    float v = 0.0f;
    if (k < Kin)             v = wih[(long long)n * Kin + k];
    else if (k < Kin + Nhh)  v = whh[(long long)n * Nhh + (k - Kin)];
    dst[idx] = f32_to_bf16_rne(v);
}

__global__ void copy_f32(const float* __restrict__ src, float* __restrict__ dst, int n) {
    int i = blockIdx.x * blockDim.x + threadIdx.x;
    if (i < n) dst[i] = src[i];
}

// ---------------------------------------------------------------------------
// Build concat activation block act[16][Kpad] (bf16) from 4 fp32 segments
// (concat order must match the pack's column order). Pad region -> 0.
// ---------------------------------------------------------------------------
__global__ void gather_act_bf16(const float* __restrict__ s0, int l0, long long st0,
                                const float* __restrict__ s1, int l1, long long st1,
                                const float* __restrict__ s2, int l2, long long st2,
                                const float* __restrict__ s3, int l3, long long st3,
                                int Kpad, uint16_t* __restrict__ dst) {
    int idx = blockIdx.x * blockDim.x + threadIdx.x;
    if (idx >= BB * Kpad) return;
    int b = idx / Kpad;
    int k = idx - b * Kpad;
    float v = 0.0f;
    if (k < l0) v = s0[(long long)b * st0 + k];
    else {
        k -= l0;
        if (k < l1) v = s1[(long long)b * st1 + k];
        else {
            k -= l1;
            if (k < l2) v = s2[(long long)b * st2 + k];
            else {
                k -= l2;
                if (k < l3) v = s3[(long long)b * st3 + k];
            }
        }
    }
    dst[idx] = f32_to_bf16_rne(v);
}

// ---------------------------------------------------------------------------
// One RNN cell step:  h' = tanh(act @ Wpacked^T + bih + bhh)
// grid.x = N/16 output tiles; 8 waves/block split K; LDS reduce; tanh; store
// fp32 state + [B,T,N] output slice.
// A-fragment follows ISA 16-bit A 16x32 layout: lane m=l&15; lanes<16 take
// K = {0..7, 16..23}, lanes>=16 take K = {8..15, 24..31} of each 32-chunk.
// ---------------------------------------------------------------------------
__global__ __launch_bounds__(256) void rnn_cell_wmma(
        const uint16_t* __restrict__ act,     // [16][Kpad] bf16
        const uint16_t* __restrict__ packW,   // [N/16][Kpad/32][32][16] bf16
        const float* __restrict__ bih, const float* __restrict__ bhh,
        float* __restrict__ state,            // [16][N] fp32
        float* __restrict__ yout,             // [B][T][N] fp32 (this tensor's base)
        int N, int Kpad, int t) {
    __shared__ float red[8 * 256];
    const int tid  = threadIdx.x;
    const int wave = tid >> 5;
    const int lane = tid & 31;
    const int nt   = blockIdx.x;
    const int Kc   = Kpad >> 5;

    const int m    = lane & 15;
    const int aoff = (lane >> 4) << 3;      // 0 or 8 bf16 elements
    const uint16_t* arow  = act + (long long)m * Kpad + aoff;
    const uint16_t* bbase = packW + ((long long)nt * Kc * 32 + lane) * 16;

    v8f acc = {0.f, 0.f, 0.f, 0.f, 0.f, 0.f, 0.f, 0.f};
    for (int kc = wave; kc < Kc; kc += 8) {
        Frag16 a, b;
        const u32x4* pa = (const u32x4*)(arow + kc * 32);
        a.q[0] = pa[0];                       // K = base + aoff + 0..7
        a.q[1] = pa[2];                       // K = base + aoff + 16..23
        const u32x4* pb = (const u32x4*)(bbase + (long long)kc * 512);
        b.q[0] = pb[0];
        b.q[1] = pb[1];
        __builtin_prefetch(bbase + (long long)(kc + 8) * 512, 0, 0);
        acc = __builtin_amdgcn_wmma_f32_16x16x32_bf16(
                false, a.v, false, b.v, (short)0, acc, false, false);
    }
    #pragma unroll
    for (int r = 0; r < 8; ++r) red[wave * 256 + lane * 8 + r] = acc[r];
    __syncthreads();

    // 256 threads == 256 elements of the 16x16 C tile (ISA C/D layout)
    const int l2 = tid & 31;
    const int r2 = tid >> 5;
    float s = 0.0f;
    #pragma unroll
    for (int w = 0; w < 8; ++w) s += red[w * 256 + l2 * 8 + r2];
    const int n  = nt * 16 + (l2 & 15);
    const int mo = r2 + ((l2 >> 4) << 3);
    float h = tanhf(s + bih[n] + bhh[n]);
    state[(long long)mo * N + n] = h;
    yout[((long long)mo * TT + t) * (long long)N + n] = h;
}

// ---------------------------------------------------------------------------
static inline unsigned cdivu(long long a, long long b) { return (unsigned)((a + b - 1) / b); }

extern "C" void kernel_launch(void* const* d_in, const int* in_sizes, int n_in,
                              void* d_out, int out_size, void* d_ws, size_t ws_size,
                              hipStream_t stream) {
    (void)in_sizes; (void)n_in; (void)out_size;

    // setup_inputs() dict order: 6 tensors + 4 params per cell x 4 cells = 22
    const float* x_on    = (const float*)d_in[0];
    const float* x_off   = (const float*)d_in[1];
    const float* h4e0    = (const float*)d_in[2];
    const float* h4i0    = (const float*)d_in[3];
    const float* h23e0   = (const float*)d_in[4];
    const float* h23i0   = (const float*)d_in[5];
    const float* w4e_ih  = (const float*)d_in[6];
    const float* w4e_hh  = (const float*)d_in[7];
    const float* b4e_ih  = (const float*)d_in[8];
    const float* b4e_hh  = (const float*)d_in[9];
    const float* w4i_ih  = (const float*)d_in[10];
    const float* w4i_hh  = (const float*)d_in[11];
    const float* b4i_ih  = (const float*)d_in[12];
    const float* b4i_hh  = (const float*)d_in[13];
    const float* w23e_ih = (const float*)d_in[14];
    const float* w23e_hh = (const float*)d_in[15];
    const float* b23e_ih = (const float*)d_in[16];
    const float* b23e_hh = (const float*)d_in[17];
    const float* w23i_ih = (const float*)d_in[18];
    const float* w23i_hh = (const float*)d_in[19];
    const float* b23i_ih = (const float*)d_in[20];
    const float* b23i_hh = (const float*)d_in[21];

    constexpr int N4E = 8000, N4I = 2000, N23E = 8000, N23I = 2000;
    // Kin + Kself, padded up to multiple of 32
    constexpr int K4E = 14016;   // 6000 + 8000  = 14000 -> 14016
    constexpr int K4I = 14016;   // 12000 + 2000 = 14000 -> 14016
    constexpr int K23E = 20000;  // 12000 + 8000 (already /32)
    constexpr int K23I = 20000;  // 18000 + 2000

    // ---- carve workspace ----
    uint8_t* base = (uint8_t*)d_ws;
    size_t cur = 0;
    auto carve = [&](size_t bytes) -> uint8_t* {
        uint8_t* p = base + cur;
        cur += (bytes + 255) & ~(size_t)255;
        return p;
    };
    uint16_t* pk4e  = (uint16_t*)carve((size_t)N4E  * K4E  * 2);
    uint16_t* pk4i  = (uint16_t*)carve((size_t)N4I  * K4I  * 2);
    uint16_t* pk23e = (uint16_t*)carve((size_t)N23E * K23E * 2);
    uint16_t* pk23i = (uint16_t*)carve((size_t)N23I * K23I * 2);
    uint16_t* a4e   = (uint16_t*)carve((size_t)BB * K4E  * 2);
    uint16_t* a4i   = (uint16_t*)carve((size_t)BB * K4I  * 2);
    uint16_t* a23e  = (uint16_t*)carve((size_t)BB * K23E * 2);
    uint16_t* a23i  = (uint16_t*)carve((size_t)BB * K23I * 2);
    float* st4e  = (float*)carve((size_t)BB * N4E  * 4);
    float* st4i  = (float*)carve((size_t)BB * N4I  * 4);
    float* st23e = (float*)carve((size_t)BB * N23E * 4);
    float* st23i = (float*)carve((size_t)BB * N23I * 4);
    if (cur > ws_size) return;   // workspace too small for bf16-packed path

    // ---- one-time weight repack (fp32 -> bf16 WMMA-B tiles) ----
    pack_weights_bf16<<<cdivu((long long)N4E * K4E, 256), 256, 0, stream>>>(
        w4e_ih, 6000, w4e_hh, N4E, N4E, K4E, pk4e);
    pack_weights_bf16<<<cdivu((long long)N4I * K4I, 256), 256, 0, stream>>>(
        w4i_ih, 12000, w4i_hh, N4I, N4I, K4I, pk4i);
    pack_weights_bf16<<<cdivu((long long)N23E * K23E, 256), 256, 0, stream>>>(
        w23e_ih, 12000, w23e_hh, N23E, N23E, K23E, pk23e);
    pack_weights_bf16<<<cdivu((long long)N23I * K23I, 256), 256, 0, stream>>>(
        w23i_ih, 18000, w23i_hh, N23I, N23I, K23I, pk23i);

    // ---- initial hidden states ----
    copy_f32<<<cdivu(BB * N4E, 256), 256, 0, stream>>>(h4e0,  st4e,  BB * N4E);
    copy_f32<<<cdivu(BB * N4I, 256), 256, 0, stream>>>(h4i0,  st4i,  BB * N4I);
    copy_f32<<<cdivu(BB * N23E, 256), 256, 0, stream>>>(h23e0, st23e, BB * N23E);
    copy_f32<<<cdivu(BB * N23I, 256), 256, 0, stream>>>(h23i0, st23i, BB * N23I);

    // ---- output slices (tuple concatenated flat: y4e, y4i, y23e, y23i) ----
    float* y4e  = (float*)d_out;
    float* y4i  = y4e  + (size_t)BB * TT * N4E;
    float* y23e = y4i  + (size_t)BB * TT * N4I;
    float* y23i = y23e + (size_t)BB * TT * N23E;

    const long long xst = (long long)TT * 2000;   // per-batch stride of x slices

    for (int t = 0; t < TT; ++t) {
        const float* xon_t  = x_on  + (size_t)t * 2000;
        const float* xoff_t = x_off + (size_t)t * 2000;

        // L4E: concat[x_on, x_off, h4i] (+ self h4e)
        gather_act_bf16<<<cdivu(BB * K4E, 256), 256, 0, stream>>>(
            xon_t, 2000, xst, xoff_t, 2000, xst,
            st4i, N4I, N4I, st4e, N4E, N4E, K4E, a4e);
        rnn_cell_wmma<<<N4E / 16, 256, 0, stream>>>(
            a4e, pk4e, b4e_ih, b4e_hh, st4e, y4e, N4E, K4E, t);

        // L4I: concat[x_on, x_off, h4e_new] (+ self h4i)
        gather_act_bf16<<<cdivu(BB * K4I, 256), 256, 0, stream>>>(
            xon_t, 2000, xst, xoff_t, 2000, xst,
            st4e, N4E, N4E, st4i, N4I, N4I, K4I, a4i);
        rnn_cell_wmma<<<N4I / 16, 256, 0, stream>>>(
            a4i, pk4i, b4i_ih, b4i_hh, st4i, y4i, N4I, K4I, t);

        // L23E: concat[h4e, h4i, h23i] (+ self h23e)
        gather_act_bf16<<<cdivu(BB * K23E, 256), 256, 0, stream>>>(
            st4e, N4E, N4E, st4i, N4I, N4I,
            st23i, N23I, N23I, st23e, N23E, N23E, K23E, a23e);
        rnn_cell_wmma<<<N23E / 16, 256, 0, stream>>>(
            a23e, pk23e, b23e_ih, b23e_hh, st23e, y23e, N23E, K23E, t);

        // L23I: concat[h4e, h4i, h23e_new] (+ self h23i)
        gather_act_bf16<<<cdivu(BB * K23I, 256), 256, 0, stream>>>(
            st4e, N4E, N4E, st4i, N4I, N4I,
            st23e, N23E, N23E, st23i, N23I, N23I, K23I, a23i);
        rnn_cell_wmma<<<N23I / 16, 256, 0, stream>>>(
            a23i, pk23i, b23i_ih, b23i_hh, st23i, y23i, N23I, K23I, t);
    }
}
